// NAT_55327768708238
// MI455X (gfx1250) — matrix operand order repaired and checked by reading, not compiled, not run on hardware
//
#include <hip/hip_runtime.h>
#include <hip/hip_bf16.h>
#include <cstdint>
#include <cstddef>

// ---------------------------------------------------------------------------
// CDNA5 (gfx1250) SwinV2 double shifted-window attention, WMMA bf16 path.
// ---------------------------------------------------------------------------

typedef __attribute__((ext_vector_type(16))) __bf16 v16bf;
typedef __attribute__((ext_vector_type(8)))  float  v8f;

union Frag16 {
    v16bf v;
    uint4 q[2];
};

__device__ __forceinline__ unsigned short f2bf(float x) {
    union { float f; unsigned u; } t; t.f = x;
    unsigned u = t.u;
    u = (u + 0x7FFFu + ((u >> 16) & 1u)) >> 16;   // round-to-nearest-even
    return (unsigned short)u;
}

__device__ __forceinline__ v8f vzero8() {
    v8f z;
#pragma unroll
    for (int r = 0; r < 8; ++r) z[r] = 0.0f;
    return z;
}

// A-operand (16x32 bf16): lane group selects K-half; two 16B chunks.
// p0 -> &row_base[kbase + lg*8]   (halves)
__device__ __forceinline__ v16bf load_frag_a(const unsigned short* p0) {
    Frag16 f;
    f.q[0] = *(const uint4*)(p0);
    f.q[1] = *(const uint4*)(p0 + 16);
    return f.v;
}

// B-operand (32x16 bf16): lane group selects K-half; two 16B chunks.
// p0 -> &col_base[kbase + lg*8]   (halves)
__device__ __forceinline__ v16bf load_frag_b(const unsigned short* p0) {
    Frag16 f;
    f.q[0] = *(const uint4*)(p0);
    f.q[1] = *(const uint4*)(p0 + 16);
    return f.v;
}

// ---------------------------------------------------------------------------
// Prep kernels
// ---------------------------------------------------------------------------

__global__ void cvt_bf16_kernel(const float* __restrict__ src,
                                unsigned short* __restrict__ dst, int n) {
    int i = blockIdx.x * blockDim.x + threadIdx.x;
    if (i < n) dst[i] = f2bf(src[i]);
}

// Continuous relative position bias MLP: tab[p][h] = 16*sigmoid(mlp(coords[p]))[h]
// coords table is 15x15 = 225 entries, hidden=512, heads=8.
__global__ void cpb_kernel(const float* __restrict__ w1,  // [512][2]
                           const float* __restrict__ b1,  // [512]
                           const float* __restrict__ w2,  // [8][512]
                           float* __restrict__ tab) {     // [225][8]
    int gid = blockIdx.x * blockDim.x + threadIdx.x;
    if (gid >= 225 * 8) return;
    int p = gid >> 3, h = gid & 7;
    int a = p / 15, b = p % 15;
    float r0 = (float)(a - 7) * (8.0f / 7.0f);
    float r1 = (float)(b - 7) * (8.0f / 7.0f);
    float t0 = copysignf(log2f(fabsf(r0) + 1.0f) * (1.0f / 3.0f), r0);
    float t1 = copysignf(log2f(fabsf(r1) + 1.0f) * (1.0f / 3.0f), r1);
    float s = 0.0f;
    for (int k = 0; k < 512; ++k) {
        float hk = t0 * w1[2 * k] + t1 * w1[2 * k + 1] + b1[k];
        hk = fmaxf(hk, 0.0f);
        s += hk * w2[h * 512 + k];
    }
    tab[p * 8 + h] = 16.0f / (1.0f + __expf(-s));
}

// ---------------------------------------------------------------------------
// Fused window-attention pass. One workgroup (8 waves) per 8x8 window.
//   grid = 2048, block = 256, dynamic LDS = LDS_TOTAL bytes
// LDS layout (bytes):
//   [0,      65536)  xw bf16[64][256]   (phase1)  /  pbuf bf16[8][64][64] (phase3)
//   [65536,  98304)  qn bf16[8][64][32] (phase2)  /  ao bf16[64][256]     (phase4)
//   [98304, 131072)  kn bf16[8][64][32]
//   [131072,163840)  vT bf16[8][32][64]
//   [163840,164096)  region int[64]
//   [164096,171296)  bias tab float[225*8]
// ---------------------------------------------------------------------------

#define LDS_TOTAL 171296

__global__ __launch_bounds__(256)
void swin_pass_kernel(const float* __restrict__ xin, float* __restrict__ xout,
                      const unsigned short* __restrict__ qkvw_bf,   // [768][256] bf16
                      const unsigned short* __restrict__ projw_bf,  // [256][256] bf16
                      const float* __restrict__ qkv_b,              // [768]
                      const float* __restrict__ proj_b,             // [256]
                      const float* __restrict__ logit_scale,        // [8]
                      const float* __restrict__ rpbtab,             // [225][8]
                      int in_chw, int out_chw, int shift) {
    extern __shared__ unsigned char smem[];
    unsigned short* xw   = (unsigned short*)(smem);            // also pbuf
    unsigned short* qn   = (unsigned short*)(smem + 65536);    // also ao
    unsigned short* kn   = (unsigned short*)(smem + 98304);
    unsigned short* vT   = (unsigned short*)(smem + 131072);
    int*            reg  = (int*)(smem + 163840);
    float*          tabL = (float*)(smem + 164096);

    const int tid  = threadIdx.x;
    const int wave = tid >> 5;
    const int lane = tid & 31;
    const int lg   = lane >> 4;     // lane group (K-half select)
    const int li   = lane & 15;     // index within group (row/col select)

    const int bt = blockIdx.x >> 6;         // image index 0..31
    const int wid = blockIdx.x & 63;        // window within image
    const int wy = wid >> 3, wx = wid & 7;

    // ---- stage window input (with roll) as bf16, region ids, bias table ----
    for (int e = tid; e < 64 * 256; e += 256) {
        int t = e >> 8, c = e & 255;
        int gh = ((wy << 3) + (t >> 3) + shift) & 63;
        int gw = ((wx << 3) + (t & 7) + shift) & 63;
        size_t adr = in_chw ? ((((size_t)bt * 256 + c) << 12) + (gh << 6) + gw)
                            : (((((size_t)bt << 12) + (gh << 6) + gw) << 8) + c);
        xw[e] = f2bf(xin[adr]);
    }
    for (int e = tid; e < 225 * 8; e += 256) tabL[e] = rpbtab[e];
    if (tid < 64) {
        int rh = (wy << 3) + (tid >> 3);
        int rw = (wx << 3) + (tid & 7);
        int bh = rh < 56 ? 0 : (rh < 60 ? 1 : 2);
        int bw = rw < 56 ? 0 : (rw < 60 ? 1 : 2);
        reg[tid] = bh * 3 + bw;
    }
    __syncthreads();

    // ---- phase 1: QKV GEMM [64,256]x[256,768], per-wave head-chunks -------
    for (int c3 = 0; c3 < 3; ++c3) {
        const int chunk = wave * 3 + c3;      // 0..7 q, 8..15 k, 16..23 v
        const int fb = chunk << 5;
        const int head = chunk & 7;
        const bool isQ = chunk < 8;
        const bool isK = (chunk >= 8) && (chunk < 16);

        v8f acc[4][2];
#pragma unroll
        for (int mt = 0; mt < 4; ++mt)
#pragma unroll
            for (int nt = 0; nt < 2; ++nt) acc[mt][nt] = vzero8();

        for (int kc = 0; kc < 8; ++kc) {
            v16bf a[4], b[2];
#pragma unroll
            for (int mt = 0; mt < 4; ++mt)
                a[mt] = load_frag_a(xw + ((mt * 16 + li) << 8) + kc * 32 + lg * 8);
#pragma unroll
            for (int nt = 0; nt < 2; ++nt) {
                int f = fb + nt * 16 + li;
                b[nt] = load_frag_b(qkvw_bf + ((size_t)f << 8) + kc * 32 + lg * 8);
            }
#pragma unroll
            for (int mt = 0; mt < 4; ++mt)
#pragma unroll
                for (int nt = 0; nt < 2; ++nt)
                    acc[mt][nt] = __builtin_amdgcn_wmma_f32_16x16x32_bf16(
                        false, a[mt], false, b[nt], (short)0, acc[mt][nt], false, false);
        }

        // bias (key bias zeroed in SwinV2)
        float bias[2];
#pragma unroll
        for (int nt = 0; nt < 2; ++nt)
            bias[nt] = isK ? 0.0f : qkv_b[fb + nt * 16 + li];
#pragma unroll
        for (int mt = 0; mt < 4; ++mt)
#pragma unroll
            for (int nt = 0; nt < 2; ++nt)
#pragma unroll
                for (int r = 0; r < 8; ++r) acc[mt][nt][r] += bias[nt];

        // cosine-attention row normalization; fold logit scale into qn
        if (isQ || isK) {
            float mult0 = 1.0f;
            if (isQ) mult0 = __expf(fminf(logit_scale[head], 4.60517018599f)); // log(100)
#pragma unroll
            for (int mt = 0; mt < 4; ++mt)
#pragma unroll
                for (int r = 0; r < 8; ++r) {
                    float ss = acc[mt][0][r] * acc[mt][0][r] +
                               acc[mt][1][r] * acc[mt][1][r];
                    ss += __shfl_xor(ss, 1, 32);
                    ss += __shfl_xor(ss, 2, 32);
                    ss += __shfl_xor(ss, 4, 32);
                    ss += __shfl_xor(ss, 8, 32);
                    float mul = mult0 / fmaxf(sqrtf(ss), 1e-12f);
                    acc[mt][0][r] *= mul;
                    acc[mt][1][r] *= mul;
                }
        }

        // store bf16: qn/kn row-major [head][64][32], v transposed [head][32][64]
#pragma unroll
        for (int mt = 0; mt < 4; ++mt)
#pragma unroll
            for (int nt = 0; nt < 2; ++nt)
#pragma unroll
                for (int r = 0; r < 8; ++r) {
                    int m = mt * 16 + r + (lg << 3);
                    int d = nt * 16 + li;
                    unsigned short bv = f2bf(acc[mt][nt][r]);
                    if (isQ)      qn[((head * 64 + m) << 5) + d] = bv;
                    else if (isK) kn[((head * 64 + m) << 5) + d] = bv;
                    else          vT[((head * 32 + d) << 6) + m] = bv;
                }
    }
    __syncthreads();

    // ---- phase 2+3: attention, one head per wave ---------------------------
    v8f o[4][2];
    {
        const int h = wave;
        v16bf a[4], b[4];
#pragma unroll
        for (int mt = 0; mt < 4; ++mt)
            a[mt] = load_frag_a(qn + ((h * 64 + mt * 16 + li) << 5) + lg * 8);
#pragma unroll
        for (int nt = 0; nt < 4; ++nt)
            b[nt] = load_frag_b(kn + ((h * 64 + nt * 16 + li) << 5) + lg * 8);

        v8f s[4][4];
#pragma unroll
        for (int mt = 0; mt < 4; ++mt)
#pragma unroll
            for (int nt = 0; nt < 4; ++nt) {
                s[mt][nt] = vzero8();
                s[mt][nt] = __builtin_amdgcn_wmma_f32_16x16x32_bf16(
                    false, a[mt], false, b[nt], (short)0, s[mt][nt], false, false);
            }

        // relative-position bias + shifted-window mask
#pragma unroll
        for (int mt = 0; mt < 4; ++mt)
#pragma unroll
            for (int nt = 0; nt < 4; ++nt)
#pragma unroll
                for (int r = 0; r < 8; ++r) {
                    int m = mt * 16 + r + (lg << 3);
                    int n = nt * 16 + li;
                    int idx = (((m >> 3) - (n >> 3) + 7) * 15) +
                              ((m & 7) - (n & 7) + 7);
                    float val = s[mt][nt][r] + tabL[idx * 8 + h];
                    if (shift && (reg[m] != reg[n])) val -= 100.0f;
                    s[mt][nt][r] = val;
                }

        // row softmax (rows live across 16 lanes x 4 N-tiles)
#pragma unroll
        for (int mt = 0; mt < 4; ++mt)
#pragma unroll
            for (int r = 0; r < 8; ++r) {
                float mx = fmaxf(fmaxf(s[mt][0][r], s[mt][1][r]),
                                 fmaxf(s[mt][2][r], s[mt][3][r]));
                mx = fmaxf(mx, __shfl_xor(mx, 1, 32));
                mx = fmaxf(mx, __shfl_xor(mx, 2, 32));
                mx = fmaxf(mx, __shfl_xor(mx, 4, 32));
                mx = fmaxf(mx, __shfl_xor(mx, 8, 32));
                float sm = 0.0f;
#pragma unroll
                for (int nt = 0; nt < 4; ++nt) {
                    float e = __expf(s[mt][nt][r] - mx);
                    s[mt][nt][r] = e;
                    sm += e;
                }
                sm += __shfl_xor(sm, 1, 32);
                sm += __shfl_xor(sm, 2, 32);
                sm += __shfl_xor(sm, 4, 32);
                sm += __shfl_xor(sm, 8, 32);
                float inv = 1.0f / sm;
#pragma unroll
                for (int nt = 0; nt < 4; ++nt) s[mt][nt][r] *= inv;
            }

        // write P (bf16) to wave-private LDS tile (reuses xw region)
        unsigned short* pb = xw + h * 4096;   // [64][64]
#pragma unroll
        for (int mt = 0; mt < 4; ++mt)
#pragma unroll
            for (int nt = 0; nt < 4; ++nt)
#pragma unroll
                for (int r = 0; r < 8; ++r) {
                    int m = mt * 16 + r + (lg << 3);
                    int n = nt * 16 + li;
                    pb[(m << 6) + n] = f2bf(s[mt][nt][r]);
                }

        // O = P @ V  (K=64 in two chunks)
#pragma unroll
        for (int mt = 0; mt < 4; ++mt)
#pragma unroll
            for (int dt = 0; dt < 2; ++dt) o[mt][dt] = vzero8();
#pragma unroll
        for (int kc = 0; kc < 2; ++kc) {
            v16bf pa[4], vb[2];
#pragma unroll
            for (int mt = 0; mt < 4; ++mt)
                pa[mt] = load_frag_a(pb + ((mt * 16 + li) << 6) + kc * 32 + lg * 8);
#pragma unroll
            for (int dt = 0; dt < 2; ++dt)
                vb[dt] = load_frag_b(vT + ((h * 32 + dt * 16 + li) << 6) + kc * 32 + lg * 8);
#pragma unroll
            for (int mt = 0; mt < 4; ++mt)
#pragma unroll
                for (int dt = 0; dt < 2; ++dt)
                    o[mt][dt] = __builtin_amdgcn_wmma_f32_16x16x32_bf16(
                        false, pa[mt], false, vb[dt], (short)0, o[mt][dt], false, false);
        }
    }
    __syncthreads();   // all waves done reading qn before reuse as ao

    // gather heads: ao[64][256] bf16 (reuses qn region)
    {
        const int h = wave;
        unsigned short* ao = qn;
#pragma unroll
        for (int mt = 0; mt < 4; ++mt)
#pragma unroll
            for (int dt = 0; dt < 2; ++dt)
#pragma unroll
                for (int r = 0; r < 8; ++r) {
                    int m = mt * 16 + r + (lg << 3);
                    int f = h * 32 + dt * 16 + li;
                    ao[(m << 8) + f] = f2bf(o[mt][dt][r]);
                }
    }
    __syncthreads();

    // ---- phase 4: output projection + residual + un-roll store ------------
    {
        const unsigned short* ao = qn;
        v8f acc[4][2];
#pragma unroll
        for (int mt = 0; mt < 4; ++mt)
#pragma unroll
            for (int nt = 0; nt < 2; ++nt) acc[mt][nt] = vzero8();

        for (int kc = 0; kc < 8; ++kc) {
            v16bf a[4], b[2];
#pragma unroll
            for (int mt = 0; mt < 4; ++mt)
                a[mt] = load_frag_a(ao + ((mt * 16 + li) << 8) + kc * 32 + lg * 8);
#pragma unroll
            for (int nt = 0; nt < 2; ++nt) {
                int f = wave * 32 + nt * 16 + li;
                b[nt] = load_frag_b(projw_bf + ((size_t)f << 8) + kc * 32 + lg * 8);
            }
#pragma unroll
            for (int mt = 0; mt < 4; ++mt)
#pragma unroll
                for (int nt = 0; nt < 2; ++nt)
                    acc[mt][nt] = __builtin_amdgcn_wmma_f32_16x16x32_bf16(
                        false, a[mt], false, b[nt], (short)0, acc[mt][nt], false, false);
        }

#pragma unroll
        for (int mt = 0; mt < 4; ++mt)
#pragma unroll
            for (int nt = 0; nt < 2; ++nt)
#pragma unroll
                for (int r = 0; r < 8; ++r) {
                    int m = mt * 16 + r + (lg << 3);
                    int f = wave * 32 + nt * 16 + li;
                    int gh = ((wy << 3) + (m >> 3) + shift) & 63;
                    int gw = ((wx << 3) + (m & 7) + shift) & 63;
                    size_t ain = in_chw
                        ? ((((size_t)bt * 256 + f) << 12) + (gh << 6) + gw)
                        : (((((size_t)bt << 12) + (gh << 6) + gw) << 8) + f);
                    size_t aout = out_chw
                        ? ((((size_t)bt * 256 + f) << 12) + (gh << 6) + gw)
                        : (((((size_t)bt << 12) + (gh << 6) + gw) << 8) + f);
                    xout[aout] = acc[mt][nt][r] + proj_b[f] + xin[ain];
                }
    }
}

// ---------------------------------------------------------------------------
// Host launcher
// ---------------------------------------------------------------------------

extern "C" void kernel_launch(void* const* d_in, const int* in_sizes, int n_in,
                              void* d_out, int out_size, void* d_ws, size_t ws_size,
                              hipStream_t stream) {
    (void)in_sizes; (void)n_in; (void)out_size; (void)ws_size;

    const float* src          = (const float*)d_in[0];
    const float* a1_qkv_w     = (const float*)d_in[1];
    const float* a1_qkv_b     = (const float*)d_in[2];
    const float* a1_proj_w    = (const float*)d_in[3];
    const float* a1_proj_b    = (const float*)d_in[4];
    const float* a1_ls        = (const float*)d_in[5];
    const float* a1_cpb_w1    = (const float*)d_in[6];
    const float* a1_cpb_b1    = (const float*)d_in[7];
    const float* a1_cpb_w2    = (const float*)d_in[8];
    const float* a2_qkv_w     = (const float*)d_in[9];
    const float* a2_qkv_b     = (const float*)d_in[10];
    const float* a2_proj_w    = (const float*)d_in[11];
    const float* a2_proj_b    = (const float*)d_in[12];
    const float* a2_ls        = (const float*)d_in[13];
    const float* a2_cpb_w1    = (const float*)d_in[14];
    const float* a2_cpb_b1    = (const float*)d_in[15];
    const float* a2_cpb_w2    = (const float*)d_in[16];

    char* ws = (char*)d_ws;
    // workspace layout
    const size_t X1_OFF     = 0;                          // float [32*64*64*256]
    const size_t QKVW1_OFF  = 134217728;                  // bf16 768*256
    const size_t PROJW1_OFF = QKVW1_OFF + 393216;         // bf16 256*256
    const size_t QKVW2_OFF  = PROJW1_OFF + 131072;
    const size_t PROJW2_OFF = QKVW2_OFF + 393216;
    const size_t TAB1_OFF   = PROJW2_OFF + 131072;        // float 225*8
    const size_t TAB2_OFF   = TAB1_OFF + 7200;

    float*          x1     = (float*)(ws + X1_OFF);
    unsigned short* qkvw1  = (unsigned short*)(ws + QKVW1_OFF);
    unsigned short* projw1 = (unsigned short*)(ws + PROJW1_OFF);
    unsigned short* qkvw2  = (unsigned short*)(ws + QKVW2_OFF);
    unsigned short* projw2 = (unsigned short*)(ws + PROJW2_OFF);
    float*          tab1   = (float*)(ws + TAB1_OFF);
    float*          tab2   = (float*)(ws + TAB2_OFF);

    // weight conversion to bf16 (read once into L2, reused by all 2048 blocks)
    cvt_bf16_kernel<<<(768 * 256 + 255) / 256, 256, 0, stream>>>(a1_qkv_w, qkvw1, 768 * 256);
    cvt_bf16_kernel<<<(256 * 256 + 255) / 256, 256, 0, stream>>>(a1_proj_w, projw1, 256 * 256);
    cvt_bf16_kernel<<<(768 * 256 + 255) / 256, 256, 0, stream>>>(a2_qkv_w, qkvw2, 768 * 256);
    cvt_bf16_kernel<<<(256 * 256 + 255) / 256, 256, 0, stream>>>(a2_proj_w, projw2, 256 * 256);

    // continuous relative position bias tables (pre-sigmoided * 16)
    cpb_kernel<<<8, 256, 0, stream>>>(a1_cpb_w1, a1_cpb_b1, a1_cpb_w2, tab1);
    cpb_kernel<<<8, 256, 0, stream>>>(a2_cpb_w1, a2_cpb_b1, a2_cpb_w2, tab2);

    // pass 1: no shift, CHW input -> HWC intermediate
    swin_pass_kernel<<<2048, 256, LDS_TOTAL, stream>>>(
        src, x1, qkvw1, projw1, a1_qkv_b, a1_proj_b, a1_ls, tab1,
        /*in_chw=*/1, /*out_chw=*/0, /*shift=*/0);

    // pass 2: shift=4 with mask, HWC input -> CHW output
    swin_pass_kernel<<<2048, 256, LDS_TOTAL, stream>>>(
        x1, (float*)d_out, qkvw2, projw2, a2_qkv_b, a2_proj_b, a2_ls, tab2,
        /*in_chw=*/0, /*out_chw=*/1, /*shift=*/4);
}
